// Attention_59253368816224
// MI455X (gfx1250) — compile-verified, matrix operand order checked by reading, not run on hardware
//
#include <hip/hip_runtime.h>
#include <hip/hip_bf16.h>
#include <stdint.h>

// ---------------------------------------------------------------------------
// Shapes (fixed by the reference)
//   B=2, S=2048, HIDDEN=2048, H=16, KV=4, D=128, M = B*S = 4096
// Workspace requirement: ~124 MB (bf16 copies + fp32 intermediates).
// ---------------------------------------------------------------------------

typedef __bf16 bf16x16 __attribute__((ext_vector_type(16)));
typedef __bf16 bf16x8  __attribute__((ext_vector_type(8)));
typedef float  f32x8   __attribute__((ext_vector_type(8)));

union FragAB { bf16x16 v; bf16x8 h[2]; };
union FragC  { f32x8 v; float f[8]; };

// Load a 16-element bf16 A/B fragment for V_WMMA_F32_16X16X32_BF16.
// rowp points to (row, k=0) of a row-major tile whose K dim is contiguous.
// Per ISA 7.12.2: lanes 0-15 take K chunks {0..7, 16..23}, lanes 16-31 take
// {8..15, 24..31}  ->  chunk0 @ half*8, chunk1 @ 16 + half*8.
__device__ inline bf16x16 load_frag(const __bf16* rowp, int half) {
  FragAB f;
  f.h[0] = *reinterpret_cast<const bf16x8*>(rowp + half * 8);
  f.h[1] = *reinterpret_cast<const bf16x8*>(rowp + 16 + half * 8);
  return f.v;
}

__device__ inline f32x8 wmma_bf16(bf16x16 a, bf16x16 b, f32x8 c) {
  // (neg_a, A, neg_b, B, c_mod, C, reuse_a, reuse_b)
  return __builtin_amdgcn_wmma_f32_16x16x32_bf16(false, a, false, b,
                                                 (short)0, c, false, false);
}

// ---------------------------------------------------------------------------
// Async global -> LDS copies (tracked by ASYNCcnt; inst offset is added to
// BOTH the LDS and global address per ISA 10 "Async Memory Instructions").
// Generic shared pointers carry the LDS byte offset in their low 32 bits.
// ---------------------------------------------------------------------------
__device__ inline void async_copy_64B(uint32_t lds, uint64_t gaddr) {
  asm volatile(
      "global_load_async_to_lds_b128 %0, %1, off\n\t"
      "global_load_async_to_lds_b128 %0, %1, off offset:16\n\t"
      "global_load_async_to_lds_b128 %0, %1, off offset:32\n\t"
      "global_load_async_to_lds_b128 %0, %1, off offset:48"
      :: "v"(lds), "v"(gaddr) : "memory");
}
__device__ inline void async_copy_32B(uint32_t lds, uint64_t gaddr) {
  asm volatile(
      "global_load_async_to_lds_b128 %0, %1, off\n\t"
      "global_load_async_to_lds_b128 %0, %1, off offset:16"
      :: "v"(lds), "v"(gaddr) : "memory");
}
__device__ inline void wait_async_le6() {
  asm volatile("s_wait_asynccnt 0x6" ::: "memory");
}
__device__ inline void wait_async_0() {
  asm volatile("s_wait_asynccnt 0x0" ::: "memory");
}
__device__ inline uint32_t lds_addr32(const void* p) {
  return (uint32_t)(uintptr_t)p;  // low 32 bits of shared-aperture address
}

// ---------------------------------------------------------------------------
// fp32 -> bf16 elementwise convert (n must be multiple of 1024)
// ---------------------------------------------------------------------------
__global__ __launch_bounds__(256) void cvt_bf16_kernel(
    const float* __restrict__ in, __bf16* __restrict__ out) {
  const size_t i = ((size_t)blockIdx.x * 256 + threadIdx.x) * 4;
  const float4 v = *reinterpret_cast<const float4*>(in + i);
  union { __bf16 b[4]; uint2 u; } cv;
  cv.b[0] = (__bf16)v.x; cv.b[1] = (__bf16)v.y;
  cv.b[2] = (__bf16)v.z; cv.b[3] = (__bf16)v.w;
  *reinterpret_cast<uint2*>(out + i) = cv.u;
}

// ---------------------------------------------------------------------------
// fp32 [R,C] -> bf16 transpose [C,R]   (R,C multiples of 32)
// ---------------------------------------------------------------------------
__global__ void transpose_bf16_kernel(const float* __restrict__ in,
                                      __bf16* __restrict__ out, int R, int C) {
  __shared__ float tile[32][33];
  const int tx = threadIdx.x, ty = threadIdx.y;
  const int cIn = blockIdx.x * 32 + tx;
  const int rBase = blockIdx.y * 32;
  for (int i = 0; i < 4; ++i)
    tile[ty + i * 8][tx] = in[(size_t)(rBase + ty + i * 8) * C + cIn];
  __syncthreads();
  const int rOut = rBase + tx;
  const int cBase = blockIdx.x * 32;
  for (int i = 0; i < 4; ++i)
    out[(size_t)(cBase + ty + i * 8) * R + rOut] = (__bf16)tile[tx][ty + i * 8];
}

// ---------------------------------------------------------------------------
// GEMM: C[M,N] = A[M,K](bf16, row-major) * Bt[N,K](bf16, i.e. B^T)  -> fp32
// Block tile 256x128, 8 waves of 64x64, BK=32.
// Tiles streamed global->LDS via GLOBAL_LOAD_ASYNC_TO_LDS_B128 (ASYNCcnt),
// double-buffered; 16 WMMA per wave per K-step.
// Each wave issues exactly 6 async-load instructions per tile, and async
// loads complete in order -> s_wait_asynccnt 6 guarantees the current
// buffer is resident before the barrier.
// ---------------------------------------------------------------------------
__global__ __launch_bounds__(256) void gemm_bf16_kernel(
    const __bf16* __restrict__ A, const __bf16* __restrict__ Bt,
    float* __restrict__ C, int M, int N, int K) {
  __shared__ __align__(16) __bf16 As[2][256 * 32];
  __shared__ __align__(16) __bf16 Bs[2][128 * 32];

  const int tid  = threadIdx.x;
  const int lane = tid & 31;
  const int wave = tid >> 5;
  const int half = lane >> 4;
  const int ln   = lane & 15;
  const int m0 = blockIdx.y * 256;
  const int n0 = blockIdx.x * 128;
  const int wm = wave >> 1;   // 0..3  -> 64-row slice
  const int wn = wave & 1;    // 0..1  -> 64-col slice

  FragC acc[4][4];
  for (int tm = 0; tm < 4; ++tm)
    for (int tn = 0; tn < 4; ++tn)
      for (int r = 0; r < 8; ++r) acc[tm][tn].f[r] = 0.f;

  // A tile 256x32: one full row (64 B) per thread.
  // B tile 128x32: half a row (32 B) per thread.
  const int aRow = tid;
  const int bRow = tid >> 1;
  const int bOff = (tid & 1) * 32;  // byte offset within B row

  const __bf16* aG = A + (size_t)(m0 + aRow) * K;
  const __bf16* bG = Bt + (size_t)(n0 + bRow) * K + (bOff >> 1);

  const uint32_t aL[2] = { lds_addr32(&As[0][aRow * 32]),
                           lds_addr32(&As[1][aRow * 32]) };
  const uint32_t bL[2] = { lds_addr32(&Bs[0][bRow * 32]) + (uint32_t)bOff,
                           lds_addr32(&Bs[1][bRow * 32]) + (uint32_t)bOff };

  // Prologue: stage K-tile 0 into buffer 0.
  async_copy_64B(aL[0], (uint64_t)(uintptr_t)aG);
  async_copy_32B(bL[0], (uint64_t)(uintptr_t)bG);

  for (int k0 = 0; k0 < K; k0 += 32) {
    const int buf = (k0 >> 5) & 1;
    if (k0 + 32 < K) {  // stream next tile into the other buffer
      async_copy_64B(aL[buf ^ 1], (uint64_t)(uintptr_t)(aG + k0 + 32));
      async_copy_32B(bL[buf ^ 1], (uint64_t)(uintptr_t)(bG + k0 + 32));
      wait_async_le6();  // 6 outstanding = only the next tile in flight
    } else {
      wait_async_0();
    }
    __syncthreads();  // every wave waited on its own slice -> tile resident

    bf16x16 af[4];
    for (int tm = 0; tm < 4; ++tm)
      af[tm] = load_frag(&As[buf][(wm * 64 + tm * 16 + ln) * 32], half);
    for (int tn = 0; tn < 4; ++tn) {
      bf16x16 bfv = load_frag(&Bs[buf][(wn * 64 + tn * 16 + ln) * 32], half);
      for (int tm = 0; tm < 4; ++tm)
        acc[tm][tn].v = wmma_bf16(af[tm], bfv, acc[tm][tn].v);
    }
    __syncthreads();  // readers done before this buffer is refilled
  }

  for (int tm = 0; tm < 4; ++tm)
    for (int tn = 0; tn < 4; ++tn)
      for (int r = 0; r < 8; ++r) {
        const int row = m0 + wm * 64 + tm * 16 + r + 8 * half;
        const int col = n0 + wn * 64 + tn * 16 + ln;
        C[(size_t)row * N + col] = acc[tm][tn].f[r];
      }
}

// ---------------------------------------------------------------------------
// RMSNorm (per-head, D=128) + RoPE, fp32 in -> bf16 out in [B, nheads, S, D]
// ---------------------------------------------------------------------------
__global__ __launch_bounds__(128) void norm_rope_kernel(
    const float* __restrict__ X, const float* __restrict__ w,
    const float* __restrict__ sinp, const float* __restrict__ cosp,
    __bf16* __restrict__ out, int nheads, int ld) {
  constexpr int S = 2048, D = 128;
  const int s = blockIdx.x, h = blockIdx.y, b = blockIdx.z;
  const int d = threadIdx.x;
  const size_t rowoff = (size_t)(b * S + s) * ld + h * D;

  const float v = X[rowoff + d];
  float v2 = v * v;
  for (int m = 16; m >= 1; m >>= 1) v2 += __shfl_xor(v2, m);
  __shared__ float red[4];
  if ((d & 31) == 0) red[d >> 5] = v2;
  __syncthreads();
  const float ms = (red[0] + red[1] + red[2] + red[3]) * (1.f / D);
  const float rs = rsqrtf(ms + 1.1920928955078125e-07f);

  const float qn = v * rs * w[d];
  const int pp = d ^ 64;
  const float pv = X[rowoff + pp] * rs * w[pp];
  const float rot = (d < 64) ? -pv : pv;
  const float o = cosp[s * D + d] * qn + sinp[s * D + d] * rot;
  out[((size_t)(b * nheads + h) * S + s) * D + d] = (__bf16)o;
}

// ---------------------------------------------------------------------------
// V pack: fp32 [B*S, KV*D] -> bf16 transposed [B, KV, D, S]
// ---------------------------------------------------------------------------
__global__ __launch_bounds__(256) void v_pack_kernel(
    const float* __restrict__ V32, __bf16* __restrict__ Vtr) {
  constexpr int S = 2048, D = 128, KV = 4;
  const int idx = blockIdx.x * 256 + threadIdx.x;
  const int s = idx & (S - 1);
  int rest = idx >> 11;
  const int d = rest & (D - 1);
  rest >>= 7;
  const int kv = rest & (KV - 1);
  const int b = rest >> 2;
  Vtr[idx] = (__bf16)V32[(size_t)(b * S + s) * (KV * D) + kv * D + d];
}

// ---------------------------------------------------------------------------
// Flash attention: 1 wave per 16-row q-tile; bf16 WMMA for QK^T and P*V,
// online softmax in registers, P re-shaped C->A layout through LDS.
// Qr: [B,H,S,D] bf16, Kr: [B,KV,S,D] bf16, Vt: [B,KV,D,S] bf16,
// Ob: [B*S, H*D] bf16 (row-major for the final projection GEMM).
// ---------------------------------------------------------------------------
__global__ __launch_bounds__(32) void flash_attn_kernel(
    const __bf16* __restrict__ Qr, const __bf16* __restrict__ Kr,
    const __bf16* __restrict__ Vt, __bf16* __restrict__ Ob) {
  constexpr int S = 2048, D = 128, H = 16, KV = 4;
  __shared__ __align__(16) __bf16 p_lds[16 * 32];

  const int lane = threadIdx.x;
  const int half = lane >> 4;
  const int ln   = lane & 15;
  const int qt = blockIdx.x, h = blockIdx.y, b = blockIdx.z;
  const int q0 = qt * 16;
  const int kvh = h >> 2;  // H/KV = 4

  const __bf16* qbase = Qr + ((size_t)(b * H + h) * S + q0) * D;
  const __bf16* kbase = Kr + (size_t)(b * KV + kvh) * S * D;
  const __bf16* vbase = Vt + (size_t)(b * KV + kvh) * D * (size_t)S;

  bf16x16 qf[4];
  for (int dc = 0; dc < 4; ++dc)
    qf[dc] = load_frag(qbase + (size_t)ln * D + dc * 32, half);

  FragC acc[8];
  for (int t = 0; t < 8; ++t)
    for (int r = 0; r < 8; ++r) acc[t].f[r] = 0.f;
  float m_r[8], l_r[8];
  for (int r = 0; r < 8; ++r) { m_r[r] = -1e30f; l_r[r] = 0.f; }

  const float sm = 0.08838834764831845f;  // 1/sqrt(128)
  const int nchunks = (q0 + 16 + 31) / 32;

  for (int c = 0; c < nchunks; ++c) {
    const int c0 = c * 32;
    if (c + 1 < nchunks) {  // prefetch next key chunk -> global_prefetch_b8
      __builtin_prefetch(kbase + (size_t)(c0 + 32 + ln) * D, 0, 1);
      __builtin_prefetch(vbase + (size_t)(ln * 16) * S + c0 + 32, 0, 1);
    }
    // ---- scores: two 16-key tiles, K-dim = D in 4 steps of 32 ----
    FragC s0, s1;
    for (int r = 0; r < 8; ++r) { s0.f[r] = 0.f; s1.f[r] = 0.f; }
    for (int dc = 0; dc < 4; ++dc) {
      bf16x16 k0f = load_frag(kbase + (size_t)(c0 + ln) * D + dc * 32, half);
      bf16x16 k1f = load_frag(kbase + (size_t)(c0 + 16 + ln) * D + dc * 32, half);
      s0.v = wmma_bf16(qf[dc], k0f, s0.v);
      s1.v = wmma_bf16(qf[dc], k1f, s1.v);
    }
    // ---- online softmax over this 32-key chunk (per row) ----
    const int key0 = c0 + ln;
    const int key1 = c0 + 16 + ln;
    float p0[8], p1[8];
    for (int r = 0; r < 8; ++r) {
      const int qrow = q0 + r + 8 * half;
      float a0 = (key0 <= qrow) ? s0.f[r] * sm : -1e30f;
      float a1 = (key1 <= qrow) ? s1.f[r] * sm : -1e30f;
      float cm = fmaxf(a0, a1);
      for (int msk = 8; msk >= 1; msk >>= 1) cm = fmaxf(cm, __shfl_xor(cm, msk));
      const float mnew = fmaxf(m_r[r], cm);
      const float sc = __expf(m_r[r] - mnew);
      const float e0 = __expf(a0 - mnew);
      const float e1 = __expf(a1 - mnew);
      float rsum = e0 + e1;
      for (int msk = 8; msk >= 1; msk >>= 1) rsum += __shfl_xor(rsum, msk);
      l_r[r] = l_r[r] * sc + rsum;
      m_r[r] = mnew;
      p0[r] = e0; p1[r] = e1;
      for (int t = 0; t < 8; ++t) acc[t].f[r] *= sc;
    }
    // ---- C-layout -> A-layout via LDS (same-wave DS ops are in-order) ----
    for (int r = 0; r < 8; ++r) {
      const int row = r + 8 * half;
      p_lds[row * 32 + ln]      = (__bf16)p0[r];
      p_lds[row * 32 + 16 + ln] = (__bf16)p1[r];
    }
    bf16x16 pf = load_frag(&p_lds[ln * 32], half);
    // ---- P * V: 8 N-tiles of D, K-dim = 32 keys ----
    for (int dt = 0; dt < 8; ++dt) {
      bf16x16 vf = load_frag(vbase + (size_t)(dt * 16 + ln) * S + c0, half);
      acc[dt].v = wmma_bf16(pf, vf, acc[dt].v);
    }
  }

  // ---- finalize: divide by l, write bf16 into [B*S, 2048] ----
  for (int r = 0; r < 8; ++r) {
    const float inv = 1.f / l_r[r];
    const int row = b * S + q0 + r + 8 * half;
    for (int dt = 0; dt < 8; ++dt)
      Ob[(size_t)row * (H * D) + h * D + dt * 16 + ln] =
          (__bf16)(acc[dt].f[r] * inv);
  }
}

// ---------------------------------------------------------------------------
// Host-side orchestration
// ---------------------------------------------------------------------------
extern "C" void kernel_launch(void* const* d_in, const int* in_sizes, int n_in,
                              void* d_out, int out_size, void* d_ws, size_t ws_size,
                              hipStream_t stream) {
  (void)in_sizes; (void)n_in; (void)out_size; (void)ws_size;
  constexpr int B = 2, S = 2048, HID = 2048, H = 16, KV = 4, D = 128;
  constexpr int M = B * S;           // 4096
  constexpr int NKV = KV * D;        // 512

  const float* x    = (const float*)d_in[0];
  const float* sinp = (const float*)d_in[1];
  const float* cosp = (const float*)d_in[2];
  const float* Wq   = (const float*)d_in[3];
  const float* Wk   = (const float*)d_in[4];
  const float* Wv   = (const float*)d_in[5];
  const float* Wo   = (const float*)d_in[6];
  const float* qw   = (const float*)d_in[7];
  const float* kw   = (const float*)d_in[8];
  float* out = (float*)d_out;

  // Workspace carve-out (256B aligned).
  char* wp = (char*)d_ws;
  auto alloc = [&](size_t bytes) {
    void* r = (void*)wp;
    wp += (bytes + 255) & ~(size_t)255;
    return r;
  };
  __bf16* xb   = (__bf16*)alloc((size_t)M * HID * 2);
  __bf16* Wqt  = (__bf16*)alloc((size_t)HID * HID * 2);
  __bf16* Wkt  = (__bf16*)alloc((size_t)NKV * HID * 2);
  __bf16* Wvt  = (__bf16*)alloc((size_t)NKV * HID * 2);
  __bf16* Wot  = (__bf16*)alloc((size_t)HID * HID * 2);
  float*  Qt   = (float*)alloc((size_t)M * HID * 4);
  float*  Kt   = (float*)alloc((size_t)M * NKV * 4);
  float*  V32  = (float*)alloc((size_t)M * NKV * 4);
  __bf16* Qr   = (__bf16*)alloc((size_t)B * H * S * D * 2);
  __bf16* Kr   = (__bf16*)alloc((size_t)B * KV * S * D * 2);
  __bf16* Vtr  = (__bf16*)alloc((size_t)B * KV * D * S * 2);
  __bf16* Ob   = (__bf16*)alloc((size_t)M * HID * 2);

  // 1) bf16 conversions / weight transposes
  cvt_bf16_kernel<<<(M * HID) / 1024, 256, 0, stream>>>(x, xb);
  transpose_bf16_kernel<<<dim3(HID / 32, HID / 32), dim3(32, 8), 0, stream>>>(Wq, Wqt, HID, HID);
  transpose_bf16_kernel<<<dim3(NKV / 32, HID / 32), dim3(32, 8), 0, stream>>>(Wk, Wkt, HID, NKV);
  transpose_bf16_kernel<<<dim3(NKV / 32, HID / 32), dim3(32, 8), 0, stream>>>(Wv, Wvt, HID, NKV);
  transpose_bf16_kernel<<<dim3(HID / 32, HID / 32), dim3(32, 8), 0, stream>>>(Wo, Wot, HID, HID);

  // 2) QKV projections (WMMA bf16 -> fp32, async-to-LDS staging)
  gemm_bf16_kernel<<<dim3(HID / 128, M / 256), 256, 0, stream>>>(xb, Wqt, Qt, M, HID, HID);
  gemm_bf16_kernel<<<dim3(NKV / 128, M / 256), 256, 0, stream>>>(xb, Wkt, Kt, M, NKV, HID);
  gemm_bf16_kernel<<<dim3(NKV / 128, M / 256), 256, 0, stream>>>(xb, Wvt, V32, M, NKV, HID);

  // 3) RMSNorm + RoPE (Q, K) and V transpose-pack
  norm_rope_kernel<<<dim3(S, H, B), 128, 0, stream>>>(Qt, qw, sinp, cosp, Qr, H, HID);
  norm_rope_kernel<<<dim3(S, KV, B), 128, 0, stream>>>(Kt, kw, sinp, cosp, Kr, KV, NKV);
  v_pack_kernel<<<(B * KV * D * S) / 256, 256, 0, stream>>>(V32, Vtr);

  // 4) Flash attention (WMMA bf16, online softmax)
  flash_attn_kernel<<<dim3(S / 16, H, B), 32, 0, stream>>>(Qr, Kr, Vtr, Ob);

  // 5) Output projection -> fp32 result
  gemm_bf16_kernel<<<dim3(HID / 128, M / 256), 256, 0, stream>>>(Ob, Wot, out, M, HID, HID);
}
